// SimpleSNN_42743514530447
// MI455X (gfx1250) — matrix operand order repaired and checked by reading, not compiled, not run on hardware
//
#include <hip/hip_runtime.h>

#define B_DIM 128
#define T_DIM 256
#define F_DIM 2500
#define H_DIM 128
#define O_DIM 2
#define KP    2528   // F_DIM padded up to a multiple of 32 (WMMA K-step)

typedef __attribute__((ext_vector_type(16))) __bf16          v16bf;
typedef __attribute__((ext_vector_type(8)))  float           v8f;
typedef __attribute__((ext_vector_type(16))) unsigned short  u16x16;
typedef __attribute__((ext_vector_type(8)))  unsigned        u32x8;
typedef __attribute__((ext_vector_type(4)))  float           f32x4;

__device__ __forceinline__ unsigned short f2bf_rne(float f) {
  unsigned u = __builtin_bit_cast(unsigned, f);
  u += 0x7FFFu + ((u >> 16) & 1u);          // round-to-nearest-even
  return (unsigned short)(u >> 16);
}
__device__ __forceinline__ float bf2f(unsigned short h) {
  unsigned u = ((unsigned)h) << 16;
  return __builtin_bit_cast(float, u);
}

// Pack two floats to two bf16 (RNE). Prefer the HW packed converter.
__device__ __forceinline__ unsigned pack_bf16_rne(float f0, float f1) {
#if __has_builtin(__builtin_amdgcn_cvt_pk_bf16_f32)
  typedef __attribute__((ext_vector_type(2))) __bf16 v2bf;
  v2bf p = __builtin_amdgcn_cvt_pk_bf16_f32(f0, f1);   // src0 -> low half
  return __builtin_bit_cast(unsigned, p);
#else
  return (unsigned)f2bf_rne(f0) | ((unsigned)f2bf_rne(f1) << 16);
#endif
}

// ---- Kernel 1: W1 (fp32, [H][F]) -> split bf16 hi/lo arrays, [H][KP], zero-padded K
__global__ __launch_bounds__(256) void snn_w1_convert(
    const float* __restrict__ W1, unsigned short* __restrict__ W1h,
    unsigned short* __restrict__ W1l) {
  int idx = blockIdx.x * 256 + threadIdx.x;
  if (idx >= H_DIM * KP) return;
  int n = idx / KP;
  int k = idx - n * KP;
  float w = (k < F_DIM) ? W1[(size_t)n * F_DIM + k] : 0.f;
  unsigned short h = f2bf_rne(w);
  W1h[idx] = h;
  W1l[idx] = f2bf_rne(w - bf2f(h));
}

// ---- One K=32 step of the split-bf16 WMMA GEMM (24 WMMAs: 8 N-tiles x 3 products)
__device__ __forceinline__ void wmma_kstep(
    const float* __restrict__ xrow, int kb, bool tail,
    const unsigned short* __restrict__ W1h,
    const unsigned short* __restrict__ W1l,
    int hlf, int l15, v8f acc[8]) {
  // A fragment: lane (hlf,l15) holds row m0+l15; elements 0..15 = K kb+hlf*16..+15
  // (contiguous K, matching the B fragment)
  float av[16];
  if (!tail) {
    const f32x4* p0 = (const f32x4*)(xrow + kb + hlf * 16);
    const f32x4* p1 = (const f32x4*)(xrow + kb + hlf * 16 + 8);
    f32x4 q0 = p0[0], q1 = p0[1], q2 = p1[0], q3 = p1[1];
#pragma unroll
    for (int e = 0; e < 4; ++e) {
      av[e] = q0[e]; av[4 + e] = q1[e]; av[8 + e] = q2[e]; av[12 + e] = q3[e];
    }
  } else {
#pragma unroll
    for (int e = 0; e < 8; ++e) {
      int k0 = kb + hlf * 16 + e;
      int k1 = kb + hlf * 16 + 8 + e;
      av[e]     = (k0 < F_DIM) ? xrow[k0] : 0.f;
      av[8 + e] = (k1 < F_DIM) ? xrow[k1] : 0.f;
    }
  }
  // split fp32 -> bf16 hi + bf16 residual, two elements per packed convert
  u32x8 hw, lw;
#pragma unroll
  for (int e = 0; e < 8; ++e) {
    float f0 = av[2 * e], f1 = av[2 * e + 1];
    unsigned h = pack_bf16_rne(f0, f1);
    float h0 = __builtin_bit_cast(float, h << 16);
    float h1 = __builtin_bit_cast(float, h & 0xffff0000u);
    hw[e] = h;
    lw[e] = pack_bf16_rne(f0 - h0, f1 - h1);
  }
  v16bf Ah = __builtin_bit_cast(v16bf, hw);
  v16bf Al = __builtin_bit_cast(v16bf, lw);

  // B fragment: lane holds column n=j*16+l15, 16 contiguous K at kb+hlf*16
  // (contiguous K, same order as the A fragment); W rows are K-contiguous so this is one 32B load.
  size_t bofs = (size_t)l15 * KP + (size_t)kb + (size_t)hlf * 16;
#pragma unroll
  for (int j = 0; j < 8; ++j) {
    v16bf Bh = __builtin_bit_cast(v16bf, *(const u16x16*)(W1h + bofs + (size_t)j * 16 * KP));
    v16bf Bl = __builtin_bit_cast(v16bf, *(const u16x16*)(W1l + bofs + (size_t)j * 16 * KP));
    acc[j] = __builtin_amdgcn_wmma_f32_16x16x32_bf16(false, Ah, false, Bh, (short)0, acc[j], false, false);
    acc[j] = __builtin_amdgcn_wmma_f32_16x16x32_bf16(false, Ah, false, Bl, (short)0, acc[j], false, false);
    acc[j] = __builtin_amdgcn_wmma_f32_16x16x32_bf16(false, Al, false, Bh, (short)0, acc[j], false, false);
  }
}

// ---- Kernel 2: cur1[m][H] = x[m][F] @ W1^T + b1, split-bf16 WMMA
// one wave per 16-row x 128-col strip: x read from HBM exactly once
__global__ __launch_bounds__(256) void snn_gemm1(
    const float* __restrict__ x, const unsigned short* __restrict__ W1h,
    const unsigned short* __restrict__ W1l, const float* __restrict__ b1,
    float* __restrict__ cur1) {
  const int lane  = threadIdx.x & 31;
  const int wv    = threadIdx.x >> 5;
  const int strip = blockIdx.x * 8 + wv;     // 2048 strips of 16 rows
  const int m0    = strip * 16;
  const int hlf   = lane >> 4;
  const int l15   = lane & 15;
  const float* xrow = x + (size_t)(m0 + l15) * F_DIM;

  v8f acc[8];
#pragma unroll
  for (int j = 0; j < 8; ++j) {
#pragma unroll
    for (int r = 0; r < 8; ++r) acc[j][r] = 0.f;
  }

  constexpr int KMAIN = (F_DIM / 32) * 32;   // 2496
#pragma unroll 2
  for (int kb = 0; kb < KMAIN; kb += 32) {
    __builtin_prefetch(xrow + kb + 256, 0, 0);   // stream x ahead
    wmma_kstep(xrow, kb, false, W1h, W1l, hlf, l15, acc);
  }
  wmma_kstep(xrow, KMAIN, true, W1h, W1l, hlf, l15, acc);  // K tail (4 valid)

  // C layout: element r of v8f = row m0 + hlf*8 + r, col = n-tile*16 + l15
#pragma unroll
  for (int j = 0; j < 8; ++j) {
    int n = j * 16 + l15;
    float bias = b1[n];
#pragma unroll
    for (int r = 0; r < 8; ++r) {
      int m = m0 + hlf * 8 + r;
      cur1[(size_t)m * H_DIM + n] = acc[j][r] + bias;
    }
  }
}

// ---- Kernel 3: LIF layer-1 scan, one thread per (b,h); in-place cur1 -> spk1
__global__ __launch_bounds__(256) void snn_scan1(float* __restrict__ cur1) {
  int idx = blockIdx.x * 256 + threadIdx.x;   // 16384 = B*H
  int b = idx >> 7;
  int h = idx & (H_DIM - 1);
  float* c = cur1 + (size_t)b * T_DIM * H_DIM + h;
  float mem = 0.f;
  for (int t = 0; t < T_DIM; ++t) {
    float cur = c[(size_t)t * H_DIM];
    float rst = (mem > 1.f) ? 1.f : 0.f;      // reset uses previous mem
    mem = 0.9f * mem + cur - rst;
    c[(size_t)t * H_DIM] = (mem > 1.f) ? 1.f : 0.f;  // spike of updated mem
  }
}

// ---- Kernel 4: layer-2 (H=128 -> O=2) + LIF scan; one wave per (b,o)
__global__ __launch_bounds__(256) void snn_layer2(
    const float* __restrict__ spk1, const float* __restrict__ W2,
    const float* __restrict__ b2, float* __restrict__ out) {
  const int lane = threadIdx.x & 31;
  const int wid  = blockIdx.x * 8 + (threadIdx.x >> 5);  // 0..255 = B*O
  const int b = wid >> 1;
  const int o = wid & 1;
  f32x4 w = *(const f32x4*)(W2 + (size_t)o * H_DIM + lane * 4);
  const float bias = b2[o];
  const float* sp = spk1 + (size_t)b * T_DIM * H_DIM + lane * 4;
  float mem = 0.f;
  for (int t = 0; t < T_DIM; ++t) {
    f32x4 s = *(const f32x4*)(sp + (size_t)t * H_DIM);
    float p = s[0] * w[0] + s[1] * w[1] + s[2] * w[2] + s[3] * w[3];
#pragma unroll
    for (int off = 16; off >= 1; off >>= 1) p += __shfl_xor(p, off, 32);
    float cur = p + bias;                     // all lanes hold full sum
    float rst = (mem > 1.f) ? 1.f : 0.f;
    mem = 0.9f * mem + cur - rst;
    float spk = (mem > 1.f) ? 1.f : 0.f;
    if (lane == 0) {
      size_t oi = ((size_t)t * B_DIM + b) * O_DIM + o;
      out[oi] = spk;                                          // spk_rec [T,B,O]
      out[(size_t)T_DIM * B_DIM * O_DIM + oi] = mem;          // mem_rec [T,B,O]
    }
  }
}

extern "C" void kernel_launch(void* const* d_in, const int* in_sizes, int n_in,
                              void* d_out, int out_size, void* d_ws, size_t ws_size,
                              hipStream_t stream) {
  const float* x  = (const float*)d_in[0];   // [B,T,F]
  const float* W1 = (const float*)d_in[1];   // [H,F]
  const float* b1 = (const float*)d_in[2];   // [H]
  const float* W2 = (const float*)d_in[3];   // [O,H]
  const float* b2 = (const float*)d_in[4];   // [O]
  float* out = (float*)d_out;

  char* ws = (char*)d_ws;
  float*          cur1 = (float*)ws;                              // 16 MiB, rows m=b*T+t
  unsigned short* W1h  = (unsigned short*)(ws + (size_t)16777216);        // 647168 B
  unsigned short* W1l  = (unsigned short*)(ws + (size_t)16777216 + 647168);

  snn_w1_convert<<<(H_DIM * KP + 255) / 256, 256, 0, stream>>>(W1, W1h, W1l);
  snn_gemm1<<<(B_DIM * T_DIM / 16) / 8, 256, 0, stream>>>(x, W1h, W1l, b1, cur1);
  snn_scan1<<<(B_DIM * H_DIM) / 256, 256, 0, stream>>>(cur1);     // in-place -> spk1
  snn_layer2<<<(B_DIM * O_DIM) / 8, 256, 0, stream>>>(cur1, W2, b2, out);
}
